// GNNCWT2D_Mk11_1sec_3H_197568496026
// MI455X (gfx1250) — compile-verified, matrix operand order checked by reading, not run on hardware
//
#include <hip/hip_runtime.h>

typedef _Float16 half_t;
typedef __attribute__((ext_vector_type(16))) _Float16 v16h;
typedef __attribute__((ext_vector_type(8)))  float    v8f;

#define B_GRAPHS 256
#define NE_NODES 19
#define NEDGE    60
#define N_NODES  (B_GRAPHS * NE_NODES)   // 4864
#define T_SAMP   25
#define F0       800
#define EPSBN    1e-5f

// ---------------------------------------------------------------------------
// Stage 0: time-average reduction  x(4864,20000) f32 -> havg(4864,800) f16
// Each block: 6400 contiguous input floats -> 256 outputs.
// Staging uses the CDNA5 async memory->LDS path (GLOBAL_LOAD_ASYNC_TO_LDS_B128,
// ASYNCcnt-tracked): one instruction per 16B, no VGPR round-trip, no ds_store.
// The single __shared__ array sits at LDS offset 0, so the LDS dest address
// VGPR is simply the byte offset into the array.
// Per-thread 25-sum afterwards (stride 25 words is conflict-free mod 64 banks
// since gcd(25,64)=1). This kernel is the bandwidth-bound bulk (~389 MB read).
// ---------------------------------------------------------------------------
__global__ __launch_bounds__(256) void k_time_avg(const float* __restrict__ x,
                                                  half_t* __restrict__ out)
{
    __shared__ __align__(16) float lds[6400];
    const int tid = threadIdx.x;
    const size_t base = (size_t)blockIdx.x * 6400;
    const uint4* src = (const uint4*)(x + base);

    for (int i = tid; i < 1600; i += 256) {
        const uint4* gp = src + i;
        const unsigned lds_off = (unsigned)i * 16u;   // byte offset, array at LDS base 0
        asm volatile("global_load_async_to_lds_b128 %0, %1, off"
                     :
                     : "v"(lds_off), "v"(gp), "v"(lds)
                     : "memory");
    }
    asm volatile("s_wait_asynccnt 0" ::: "memory");
    __syncthreads();

    float s = 0.f;
    const int o = tid * T_SAMP;
#pragma unroll
    for (int t = 0; t < T_SAMP; ++t) s += lds[o + t];
    out[(size_t)blockIdx.x * 256 + tid] = (half_t)(s * (1.0f / 25.0f));
}

// ---------------------------------------------------------------------------
// f32 -> f16 weight conversion (tiny, once per launch, deterministic)
// ---------------------------------------------------------------------------
__global__ void k_f32_to_f16(const float* __restrict__ a, half_t* __restrict__ o, int n)
{
    int i = blockIdx.x * blockDim.x + threadIdx.x;
    if (i < n) o[i] = (half_t)a[i];
}

// ---------------------------------------------------------------------------
// WMMA fragment loader: two contiguous 8-half (16B) chunks per lane.
// k(e) = (e<8 ? e : e+8) + (lane<16 ? 0 : 8)   [ISA 7.12.2, 16-bit A 16x32]
// ---------------------------------------------------------------------------
__device__ __forceinline__ v16h load_frag16(const half_t* __restrict__ p)
{
    union { uint4 u; half_t h[8]; } lo, hi;
    lo.u = *(const uint4*)(p);
    hi.u = *(const uint4*)(p + 16);
    v16h r;
#pragma unroll
    for (int i = 0; i < 8; ++i) { r[i] = lo.h[i]; r[8 + i] = hi.h[i]; }
    return r;
}

// ---------------------------------------------------------------------------
// Encoder GEMM: out(M=4864,N) = relu(A(M,K) @ W(N,K)^T + b) * bnScale[row%19] + bnBias
// Block = one 16-row M-tile (A tile staged in LDS), 8 waves cover N/16 tiles.
// Emits v_wmma_f32_16x16x32_f16. Unroll capped at 2 to keep VGPRs < 256
// (full unroll preloaded all 25 A-fragments -> s_set_vgpr_msb juggling and
// ~1 wave/SIMD; more resident waves hide the B-stream latency better).
// ---------------------------------------------------------------------------
template<int K, int N, bool OUT_F16>
__global__ __launch_bounds__(256) void enc_gemm(
    const half_t* __restrict__ A, const half_t* __restrict__ W,
    const float* __restrict__ lb,
    const float* __restrict__ bng, const float* __restrict__ bnb,
    const float* __restrict__ bnrm, const float* __restrict__ bnrv,
    void* __restrict__ outp)
{
    __shared__ __align__(16) half_t sA[16 * K];
    const int tid = threadIdx.x;
    const int mtile = blockIdx.x;

    { // stage 16 consecutive A rows (contiguous in memory) into LDS
        const uint4* src = (const uint4*)(A + (size_t)mtile * 16 * K);
        uint4* dstv = (uint4*)sA;
        const int nvec = (16 * K) / 8;           // halves per uint4 = 8
        for (int i = tid; i < nvec; i += 256) dstv[i] = src[i];
    }
    __syncthreads();

    const int wave = tid >> 5;
    const int lane = tid & 31;
    const int mloc = lane & 15;
    const int khalf = (lane < 16) ? 0 : 8;

    for (int nt = wave; nt < N / 16; nt += 8) {
        v8f acc = {};
        const int n = nt * 16 + mloc;            // this lane's output column
        const half_t* wrow = W + (size_t)n * K;  // B frag: column n of W^T = row n of W
        const half_t* arow = sA + (size_t)mloc * K;
        if (nt + 8 < N / 16)                     // hint next tile's W rows into cache
            __builtin_prefetch(wrow + (size_t)128 * K, 0, 1);
#pragma unroll 2
        for (int ks = 0; ks < K / 32; ++ks) {
            const int kb = ks * 32 + khalf;
            v16h a = load_frag16(arow + kb);
            v16h b = load_frag16(wrow + kb);
            acc = __builtin_amdgcn_wmma_f32_16x16x32_f16(
                false, a, false, b, (short)0, acc, false, false);
        }
        const float bias_n = lb[n];
#pragma unroll
        for (int r = 0; r < 8; ++r) {
            const int m = r + ((lane < 16) ? 0 : 8);
            const int row = mtile * 16 + m;
            float v = acc[r] + bias_n;
            v = v > 0.f ? v : 0.f;
            const int ne = row % NE_NODES;       // per-electrode BN channel
            const float sc = bng[ne] * rsqrtf(bnrv[ne] + EPSBN);
            v = v * sc + (bnb[ne] - bnrm[ne] * sc);
            if (OUT_F16) ((half_t*)outp)[(size_t)row * N + n] = (half_t)v;
            else         ((float*)outp)[(size_t)row * N + n] = v;
        }
    }
}

// ---------------------------------------------------------------------------
// Fused graph stage: block = one graph (19 nodes, 60 edges), all in LDS.
// GraphConv1 + BN6 -> GraphConv2 + BN7 -> max-pool -> lin5 -> 3 heads.
// ---------------------------------------------------------------------------
__global__ __launch_bounds__(128) void k_graph(
    const float* __restrict__ h64in, const int* __restrict__ eidx,
    const float* __restrict__ ew1,
    const float* __restrict__ w1rel, const float* __restrict__ b1rel, const float* __restrict__ w1root,
    const float* __restrict__ g6, const float* __restrict__ b6,
    const float* __restrict__ rm6, const float* __restrict__ rv6,
    const float* __restrict__ ew2,
    const float* __restrict__ w2rel, const float* __restrict__ b2rel, const float* __restrict__ w2root,
    const float* __restrict__ g7, const float* __restrict__ b7,
    const float* __restrict__ rm7, const float* __restrict__ rv7,
    const float* __restrict__ l5w, const float* __restrict__ l5b,
    const float* __restrict__ hmw, const float* __restrict__ hmb,
    const float* __restrict__ hbw, const float* __restrict__ hbb,
    const float* __restrict__ hdw, const float* __restrict__ hdb,
    float* __restrict__ outp)
{
    __shared__ float sh64[NE_NODES * 64];
    __shared__ float sagg[NE_NODES * 64];
    __shared__ float sh32[NE_NODES * 32];
    __shared__ float sout[NE_NODES * 32];
    __shared__ float sz[32], sz2[32];

    const int g = blockIdx.x, tid = threadIdx.x;
    const int node0 = g * NE_NODES;
    const int eb = g * NEDGE;

    for (int i = tid; i < NE_NODES * 64; i += 128) {
        sh64[i] = h64in[(size_t)node0 * 64 + i];
        sagg[i] = 0.f;
    }
    __syncthreads();

    // conv1 message scatter-add (LDS float atomics)
    for (int i = tid; i < NEDGE * 64; i += 128) {
        const int e = i >> 6, f = i & 63;
        const int sN = eidx[eb + e] - node0;
        const int dN = eidx[B_GRAPHS * NEDGE + eb + e] - node0;
        atomicAdd(&sagg[dN * 64 + f], sh64[sN * 64 + f] * ew1[e]);
    }
    __syncthreads();

    // conv1 projection + relu + BN6 (per-feature)
    for (int o = tid; o < NE_NODES * 32; o += 128) {
        const int n = o >> 5, j = o & 31;
        float sa = b1rel[j], sr = 0.f;
        for (int k = 0; k < 64; ++k) {
            sa += sagg[n * 64 + k] * w1rel[j * 64 + k];
            sr += sh64[n * 64 + k] * w1root[j * 64 + k];
        }
        float v = sa + sr; v = v > 0.f ? v : 0.f;
        const float sc = g6[j] * rsqrtf(rv6[j] + EPSBN);
        sh32[o] = v * sc + (b6[j] - rm6[j] * sc);
    }
    __syncthreads();
    for (int i = tid; i < NE_NODES * 32; i += 128) sagg[i] = 0.f;
    __syncthreads();

    // conv2 message scatter-add
    for (int i = tid; i < NEDGE * 32; i += 128) {
        const int e = i >> 5, f = i & 31;
        const int sN = eidx[eb + e] - node0;
        const int dN = eidx[B_GRAPHS * NEDGE + eb + e] - node0;
        atomicAdd(&sagg[dN * 32 + f], sh32[sN * 32 + f] * ew2[e]);
    }
    __syncthreads();

    // conv2 projection + relu + BN7
    for (int o = tid; o < NE_NODES * 32; o += 128) {
        const int n = o >> 5, j = o & 31;
        float sa = b2rel[j], sr = 0.f;
        for (int k = 0; k < 32; ++k) {
            sa += sagg[n * 32 + k] * w2rel[j * 32 + k];
            sr += sh32[n * 32 + k] * w2root[j * 32 + k];
        }
        float v = sa + sr; v = v > 0.f ? v : 0.f;
        const float sc = g7[j] * rsqrtf(rv7[j] + EPSBN);
        sout[o] = v * sc + (b7[j] - rm7[j] * sc);
    }
    __syncthreads();

    if (tid < 32) { // global max pool over 19 nodes
        float m = sout[tid];
        for (int n = 1; n < NE_NODES; ++n) m = fmaxf(m, sout[n * 32 + tid]);
        sz[tid] = m;
    }
    __syncthreads();
    if (tid < 32) { // lin5 + relu
        float v = l5b[tid];
        for (int k = 0; k < 32; ++k) v += l5w[tid * 32 + k] * sz[k];
        sz2[tid] = v > 0.f ? v : 0.f;
    }
    __syncthreads();
    if (tid < 3) {
        float v = hmb[tid];
        for (int k = 0; k < 32; ++k) v += hmw[tid * 32 + k] * sz2[k];
        outp[g * 3 + tid] = v;
    } else if (tid < 5) {
        const int j = tid - 3;
        float v = hbb[j];
        for (int k = 0; k < 32; ++k) v += hbw[j * 32 + k] * sz2[k];
        outp[B_GRAPHS * 3 + g * 2 + j] = v;
    } else if (tid < 7) {
        const int j = tid - 5;
        float v = hdb[j];
        for (int k = 0; k < 32; ++k) v += hdw[j * 32 + k] * sz2[k];
        outp[B_GRAPHS * 5 + g * 2 + j] = v;
    }
}

// ---------------------------------------------------------------------------
extern "C" void kernel_launch(void* const* d_in, const int* in_sizes, int n_in,
                              void* d_out, int out_size, void* d_ws, size_t ws_size,
                              hipStream_t stream)
{
    (void)in_sizes; (void)n_in; (void)out_size; (void)ws_size;

    const float* x      = (const float*)d_in[0];
    const int*   eidx   = (const int*)d_in[1];
    const float* lin2_w = (const float*)d_in[3];
    const float* lin2_b = (const float*)d_in[4];
    const float* bn3_g  = (const float*)d_in[5];
    const float* bn3_b  = (const float*)d_in[6];
    const float* bn3_rm = (const float*)d_in[7];
    const float* bn3_rv = (const float*)d_in[8];
    const float* lin3_w = (const float*)d_in[9];
    const float* lin3_b = (const float*)d_in[10];
    const float* bn4_g  = (const float*)d_in[11];
    const float* bn4_b  = (const float*)d_in[12];
    const float* bn4_rm = (const float*)d_in[13];
    const float* bn4_rv = (const float*)d_in[14];
    const float* lin4_w = (const float*)d_in[15];
    const float* lin4_b = (const float*)d_in[16];
    const float* bn5_g  = (const float*)d_in[17];
    const float* bn5_b  = (const float*)d_in[18];
    const float* bn5_rm = (const float*)d_in[19];
    const float* bn5_rv = (const float*)d_in[20];
    const float* ew1    = (const float*)d_in[21];
    const float* w1rel  = (const float*)d_in[22];
    const float* b1rel  = (const float*)d_in[23];
    const float* w1root = (const float*)d_in[24];
    const float* bn6_g  = (const float*)d_in[25];
    const float* bn6_b  = (const float*)d_in[26];
    const float* bn6_rm = (const float*)d_in[27];
    const float* bn6_rv = (const float*)d_in[28];
    const float* ew2    = (const float*)d_in[29];
    const float* w2rel  = (const float*)d_in[30];
    const float* b2rel  = (const float*)d_in[31];
    const float* w2root = (const float*)d_in[32];
    const float* bn7_g  = (const float*)d_in[33];
    const float* bn7_b  = (const float*)d_in[34];
    const float* bn7_rm = (const float*)d_in[35];
    const float* bn7_rv = (const float*)d_in[36];
    const float* lin5_w = (const float*)d_in[37];
    const float* lin5_b = (const float*)d_in[38];
    const float* hm_w   = (const float*)d_in[39];
    const float* hm_b   = (const float*)d_in[40];
    const float* hb_w   = (const float*)d_in[41];
    const float* hb_b   = (const float*)d_in[42];
    const float* hd_w   = (const float*)d_in[43];
    const float* hd_b   = (const float*)d_in[44];

    float* out = (float*)d_out;

    // workspace layout (bytes)
    char* ws = (char*)d_ws;
    half_t* havg = (half_t*)(ws + 0);            // 4864*800*2  = 7,782,400
    half_t* w2h  = (half_t*)(ws + 7782400);      // 256*800*2   =   409,600
    half_t* w3h  = (half_t*)(ws + 8192000);      // 128*256*2   =    65,536
    half_t* w4h  = (half_t*)(ws + 8257536);      //  64*128*2   =    16,384
    half_t* h1   = (half_t*)(ws + 8273920);      // 4864*256*2  = 2,490,368
    half_t* h2   = (half_t*)(ws + 10764288);     // 4864*128*2  = 1,245,184
    float*  h3   = (float*) (ws + 12009472);     // 4864*64*4   = 1,245,184

    // 0) time-average (the 389 MB bandwidth-bound stage, async->LDS staging)
    k_time_avg<<<(N_NODES * F0) / 256, 256, 0, stream>>>(x, havg);

    // weight conversions for WMMA
    k_f32_to_f16<<<(256 * 800 + 255) / 256, 256, 0, stream>>>(lin2_w, w2h, 256 * 800);
    k_f32_to_f16<<<(128 * 256 + 255) / 256, 256, 0, stream>>>(lin3_w, w3h, 128 * 256);
    k_f32_to_f16<<<(64 * 128 + 255) / 256, 256, 0, stream>>>(lin4_w, w4h, 64 * 128);

    // 1-3) encoder GEMMs with fused bias+relu+BN (v_wmma_f32_16x16x32_f16)
    enc_gemm<800, 256, true ><<<N_NODES / 16, 256, 0, stream>>>(
        havg, w2h, lin2_b, bn3_g, bn3_b, bn3_rm, bn3_rv, (void*)h1);
    enc_gemm<256, 128, true ><<<N_NODES / 16, 256, 0, stream>>>(
        h1, w3h, lin3_b, bn4_g, bn4_b, bn4_rm, bn4_rv, (void*)h2);
    enc_gemm<128, 64, false><<<N_NODES / 16, 256, 0, stream>>>(
        h2, w4h, lin4_b, bn5_g, bn5_b, bn5_rm, bn5_rv, (void*)h3);

    // 4) fused graph stage + heads
    k_graph<<<B_GRAPHS, 128, 0, stream>>>(
        h3, eidx,
        ew1, w1rel, b1rel, w1root, bn6_g, bn6_b, bn6_rm, bn6_rv,
        ew2, w2rel, b2rel, w2root, bn7_g, bn7_b, bn7_rm, bn7_rv,
        lin5_w, lin5_b, hm_w, hm_b, hb_w, hb_b, hd_w, hd_b,
        out);
}
